// MatryoshkaDilatedCostVolumeList_73632919323063
// MI455X (gfx1250) — matrix operand order repaired and checked by reading, not compiled, not run on hardware
//
#include <hip/hip_runtime.h>
#include <hip/hip_bf16.h>

// ---- types for WMMA ----
typedef __attribute__((ext_vector_type(16))) _Float16 v16h;
typedef __attribute__((ext_vector_type(8)))  _Float16 half8;
typedef __attribute__((ext_vector_type(8)))  float    v8f;

#define C_DIM 256
#define HO 48
#define WO 64
#define H0 192
#define W0 256
#define EPSN 1e-9f
#define SQ 264   // padded LDS row stride (halves) for [col][chan] tiles (16B aligned)

// 39 distinct signed row offsets (u-4)*d over d in {1,2,3,5,9,16}, u in 0..8
__constant__ int OFFS[39] = {
  -64,-48,-36,-32,-27,-20,-18,-16,-15,-12,-10,-9,-8,-6,-5,-4,-3,-2,-1,
    0,  1,  2,  3,  4,  5,  6,  8,  9, 10, 12, 15, 16, 18, 20, 27, 32, 36, 48, 64 };

// -------- inverse L2 norm over channels at (optionally strided) positions --------
__global__ void inv_norm_kernel(const float* __restrict__ src, float* __restrict__ dst,
                                int B, int H, int W, int Ho, int Wo, int stride) {
  int idx = blockIdx.x * blockDim.x + threadIdx.x;
  int total = B * Ho * Wo;
  if (idx >= total) return;
  int j = idx % Wo; int t = idx / Wo;
  int i = t % Ho;   int b = t / Ho;
  const float* p = src + (((size_t)b * C_DIM) * H + (size_t)i * stride) * W + (size_t)j * stride;
  size_t cstride = (size_t)H * W;
  float s = 0.f;
  #pragma unroll 4
  for (int c = 0; c < C_DIM; ++c) { float v = p[c * cstride]; s += v * v; }
  dst[idx] = 1.0f / (sqrtf(s) + EPSN);
}

// -------- scale 1: stride 1, dilations {1,2,3,5,9,16}, channels 1..6 of out --------
__global__ __launch_bounds__(256) void corr_scale1_kernel(
    const float* __restrict__ x1, const float* __restrict__ x2,
    const float* __restrict__ invq, const float* __restrict__ invk,
    float* __restrict__ out) {
  __shared__ __align__(16) _Float16 lq[WO * SQ];   // [j][c] f16
  __shared__ __align__(16) _Float16 lk[WO * SQ];
  __shared__ float sinvq[WO];
  __shared__ float sinvk[WO];

  int task = blockIdx.x;
  int offi = task % 39; int t = task / 39;
  int i = t % HO;       int b = t / HO;
  int off = OFFS[offi];
  int i2 = i + off;
  if (i2 < 0 || i2 >= HO) return;              // whole row pair OOB -> outputs stay zero

  const float* qsrc = x1 + (((size_t)b * C_DIM) * HO + i)  * WO;
  const float* ksrc = x2 + (((size_t)b * C_DIM) * HO + i2) * WO;
  for (int e = threadIdx.x; e < C_DIM * WO; e += 256) {
    int c = e >> 6, j = e & 63;
    lq[j * SQ + c] = (_Float16)qsrc[(size_t)c * (HO * WO) + j];
    lk[j * SQ + c] = (_Float16)ksrc[(size_t)c * (HO * WO) + j];
  }
  if (threadIdx.x < WO) {
    sinvq[threadIdx.x] = invq[((size_t)b * HO + i)  * WO + threadIdx.x];
    sinvk[threadIdx.x] = invk[((size_t)b * HO + i2) * WO + threadIdx.x];
  }
  __syncthreads();

  int lane = threadIdx.x & 31;
  int wave = threadIdx.x >> 5;
  int mlo  = lane & 15;
  const int DILS[6] = {1, 2, 3, 5, 9, 16};

  // 16 Gram tiles over a 4x4 grid; each wave owns 2 tiles sharing the A fragment:
  // tm = wave>>1, tn = 2*(wave&1) and 2*(wave&1)+1  -> 2 WMMAs per B-load pair.
  int tm  = wave >> 1;
  int tn0 = (wave & 1) * 2;
  int jA  = tm * 16 + mlo;                     // A-matrix row (query column)
  int jB0 = tn0 * 16 + mlo;                    // B-matrix col (key column), tile 0
  int jB1 = jB0 + 16;                          // tile 1

  v8f acc[2] = {{}, {}};
  #pragma unroll
  for (int kb = 0; kb < 8; ++kb) {
    v16h a, b0, b1;
    int abase = kb * 32 + ((lane < 16) ? 0 : 8);
    ((half8*)&a)[0] = *(const half8*)&lq[jA * SQ + abase];
    ((half8*)&a)[1] = *(const half8*)&lq[jA * SQ + abase + 16];
    int bbase = kb * 32 + ((lane < 16) ? 0 : 8);
    ((half8*)&b0)[0] = *(const half8*)&lk[jB0 * SQ + bbase];
    ((half8*)&b0)[1] = *(const half8*)&lk[jB0 * SQ + bbase + 16];
    ((half8*)&b1)[0] = *(const half8*)&lk[jB1 * SQ + bbase];
    ((half8*)&b1)[1] = *(const half8*)&lk[jB1 * SQ + bbase + 16];
    acc[0] = __builtin_amdgcn_wmma_f32_16x16x32_f16(false, a, false, b0,
                                                    (short)0, acc[0], false, false);
    acc[1] = __builtin_amdgcn_wmma_f32_16x16x32_f16(false, a, false, b1,
                                                    (short)0, acc[1], false, false);
  }

  // scatter: D[m,n] -> j = tm*16+m (query col), j2/n (key col); voff = n - j
  #pragma unroll
  for (int s = 0; s < 2; ++s) {
    int n = (tn0 + s) * 16 + mlo;
    float ik = sinvk[n];
    #pragma unroll
    for (int r = 0; r < 8; ++r) {
      int m = r + ((lane < 16) ? 0 : 8);
      int j = tm * 16 + m;
      int voff = n - j;
      float val = acc[s][r] * sinvq[j] * ik;
      #pragma unroll
      for (int dd = 0; dd < 6; ++dd) {
        int d = DILS[dd];
        if (off % d) continue;
        int u = off / d + 4;
        if (u < 0 || u > 8) continue;
        if (voff % d) continue;
        int v = voff / d + 4;
        if (v < 0 || v > 8) continue;
        out[(((((size_t)b * 7 + 1 + dd) * 9 + u) * 9 + v) * HO + i) * WO + j] = val;
      }
    }
  }
}

// -------- scale 0: stride 4, dilation 1, channel 0 of out --------
__global__ __launch_bounds__(256) void corr_scale0_kernel(
    const float* __restrict__ x1, const float* __restrict__ x2,
    const float* __restrict__ invq, const float* __restrict__ invk,
    float* __restrict__ out) {
  __shared__ __align__(16) _Float16 lq[16 * SQ];   // 16 query cols x 256 ch
  __shared__ __align__(16) _Float16 lk[96 * SQ];   // 96-col key band x 256 ch
  __shared__ float sinvq[16];
  __shared__ float sinvk[96];

  int task = blockIdx.x;
  int jt  = task & 3;  int t = task >> 2;
  int dui = t % 9;     t /= 9;
  int i   = t % HO;    int b = t / HO;
  int du  = dui - 4;
  int y2  = 4 * i + du;
  if (y2 < 0 || y2 >= H0) return;
  int j0 = jt * 16;
  int xbase = 64 * jt - 16;

  const float* qsrc = x1 + (((size_t)b * C_DIM) * H0 + 4 * i) * W0;
  for (int e = threadIdx.x; e < 16 * C_DIM; e += 256) {
    int c = e >> 4, jj = e & 15;
    lq[jj * SQ + c] = (_Float16)qsrc[(size_t)c * (H0 * W0) + 4 * (j0 + jj)];
  }
  const float* ksrc = x2 + (((size_t)b * C_DIM) * H0 + y2) * W0;
  for (int e = threadIdx.x; e < 96 * C_DIM; e += 256) {
    int lx = e % 96; int c = e / 96;
    int x = xbase + lx;
    float v = (x >= 0 && x < W0) ? ksrc[(size_t)c * (H0 * W0) + x] : 0.0f;
    lk[lx * SQ + c] = (_Float16)v;
  }
  if (threadIdx.x < 16) {
    sinvq[threadIdx.x] = invq[((size_t)b * HO + i) * WO + j0 + threadIdx.x];
  }
  if (threadIdx.x < 96) {
    int x = xbase + threadIdx.x;
    sinvk[threadIdx.x] = (x >= 0 && x < W0)
        ? invk[((size_t)b * H0 + y2) * W0 + x] : 0.0f;
  }
  __syncthreads();

  int lane = threadIdx.x & 31;
  int wave = threadIdx.x >> 5;
  if (wave >= 6) return;                       // wave-uniform; 6 band tiles
  int mlo = lane & 15;
  int lxB = wave * 16 + mlo;
  v8f acc = {};
  #pragma unroll
  for (int kb = 0; kb < 8; ++kb) {
    v16h a, bf;
    int abase = kb * 32 + ((lane < 16) ? 0 : 8);
    ((half8*)&a)[0] = *(const half8*)&lq[mlo * SQ + abase];
    ((half8*)&a)[1] = *(const half8*)&lq[mlo * SQ + abase + 16];
    int bbase = kb * 32 + ((lane < 16) ? 0 : 8);
    ((half8*)&bf)[0] = *(const half8*)&lk[lxB * SQ + bbase];
    ((half8*)&bf)[1] = *(const half8*)&lk[lxB * SQ + bbase + 16];
    acc = __builtin_amdgcn_wmma_f32_16x16x32_f16(false, a, false, bf,
                                                 (short)0, acc, false, false);
  }
  int x = xbase + wave * 16 + mlo;             // key column (full res)
  if (x < 0 || x >= W0) return;
  float ik = sinvk[wave * 16 + mlo];
  #pragma unroll
  for (int r = 0; r < 8; ++r) {
    int m = r + ((lane < 16) ? 0 : 8);
    int j = j0 + m;
    int dv = x - 4 * j;
    if (dv < -4 || dv > 4) continue;
    float val = acc[r] * sinvq[m] * ik;
    out[(((((size_t)b * 7 + 0) * 9 + dui) * 9 + (dv + 4)) * HO + i) * WO + j] = val;
  }
}

extern "C" void kernel_launch(void* const* d_in, const int* in_sizes, int n_in,
                              void* d_out, int out_size, void* d_ws, size_t ws_size,
                              hipStream_t stream) {
  const float* x1_0 = (const float*)d_in[0];   // (4,256,192,256)
  const float* x1_1 = (const float*)d_in[1];   // (4,256,48,64)
  const float* x2_0 = (const float*)d_in[2];   // (4,256,192,256)
  const float* x2_1 = (const float*)d_in[3];   // (4,256,48,64)
  float* out = (float*)d_out;
  float* ws  = (float*)d_ws;

  float* invk0 = ws;                 // 4*192*256 = 196608
  float* invq0 = ws + 196608;        // 4*48*64   = 12288 (strided x1_0 positions)
  float* invq1 = ws + 208896;        // 12288
  float* invk1 = ws + 221184;        // 12288

  hipMemsetAsync(d_out, 0, (size_t)out_size * sizeof(float), stream);

  inv_norm_kernel<<<(4*192*256 + 255) / 256, 256, 0, stream>>>(x2_0, invk0, 4, 192, 256, 192, 256, 1);
  inv_norm_kernel<<<(4*48*64   + 255) / 256, 256, 0, stream>>>(x1_0, invq0, 4, 192, 256,  48,  64, 4);
  inv_norm_kernel<<<(4*48*64   + 255) / 256, 256, 0, stream>>>(x1_1, invq1, 4,  48,  64,  48,  64, 1);
  inv_norm_kernel<<<(4*48*64   + 255) / 256, 256, 0, stream>>>(x2_1, invk1, 4,  48,  64,  48,  64, 1);

  corr_scale0_kernel<<<4 * 48 * 9 * 4, 256, 0, stream>>>(x1_0, x2_0, invq0, invk0, out);
  corr_scale1_kernel<<<4 * 48 * 39,    256, 0, stream>>>(x1_1, x2_1, invq1, invk1, out);
}
